// Transformer_23106924053196
// MI455X (gfx1250) — compile-verified, matrix operand order checked by reading, not run on hardware
//
#include <hip/hip_runtime.h>
#include <cstdint>
#include <cstddef>

// ---------------------------------------------------------------------------
// Transformer forward (6 layers) for MI455X / gfx1250.
// GEMMs: v_wmma_f32_16x16x32_bf16, double-buffered LDS tiles fed by
// GLOBAL_LOAD_ASYNC_TO_LDS_B128 (ASYNCcnt DMA).  Attention Q/K tiles staged
// by the Tensor Data Mover (tensor_load_to_lds, TENSORcnt).  Flash attention
// with online softmax.  Hidden state h lives in d_out (fp32 residual stream).
// ---------------------------------------------------------------------------

#define DEV __device__ __forceinline__

typedef __attribute__((ext_vector_type(16))) __bf16 v16bf;
typedef __attribute__((ext_vector_type(8)))  float  v8f;
typedef __attribute__((ext_vector_type(4)))  unsigned u32x4;
typedef __attribute__((ext_vector_type(8)))  unsigned u32x8;

union FragBF { v16bf v; uint4 q[2]; unsigned short s[16]; };
union U32x8  { unsigned w[8]; u32x8 v; };
union U32x4  { unsigned w[4]; u32x4 v; };

static constexpr int TB    = 8;      // batch
static constexpr int TN    = 1024;   // seq len
static constexpr int TDIM  = 768;
static constexpr int TDEPTH= 6;
static constexpr int THEADS= 12;
static constexpr int TDH   = 64;
static constexpr int TMLP  = 3072;
static constexpr int TOK   = TB * TN;           // 8192 tokens
static constexpr int QKVW  = 3 * TDIM;          // 2304

DEV unsigned short f2bf(float f) {
  union { float f; unsigned u; } x; x.f = f;
  unsigned r = x.u + 0x7fffu + ((x.u >> 16) & 1u);   // round-to-nearest-even
  return (unsigned short)(r >> 16);
}

DEV float gelu_tanh(float x) {
  const float c = 0.7978845608028654f;               // sqrt(2/pi)
  return 0.5f * x * (1.0f + tanhf(c * (x + 0.044715f * x * x * x)));
}

// Load a 16x32 bf16 fragment (A-layout per CDNA5 ISA 7.12.2) from an
// [rows x stride] row-major bf16 tile in LDS.
DEV v16bf load_frag16x32(const unsigned short* base, int stride, int k0) {
  int lane = threadIdx.x & 31;
  int r    = lane & 15;
  int kofs = k0 + ((lane >> 4) << 3);
  const unsigned short* p = base + r * stride + kofs;
  FragBF f;
  f.q[0] = *(const uint4*)(p);
  f.q[1] = *(const uint4*)(p + 16);
  return f.v;
}

DEV v8f wmma_bf16(v16bf a, v16bf b, v8f c) {
  return __builtin_amdgcn_wmma_f32_16x16x32_bf16(
      /*neg_a=*/false, a, /*neg_b=*/false, b,
      /*c_mod=*/(short)0, c, /*reuse_a=*/false, /*reuse_b=*/false);
}

// gfx1250 async DMA: copy 16B/lane from global to LDS, tracked by ASYNCcnt.
DEV void async_lds_b128(unsigned lds_byte, const unsigned short* g) {
  asm volatile("global_load_async_to_lds_b128 %0, %1, off"
               :: "v"(lds_byte), "v"((unsigned long long)(uintptr_t)g)
               : "memory");
}

// gfx1250 Tensor Data Mover: DMA a 2D bf16 tile (tile_d0 x tile_d1 elements,
// row stride = stride0 elements) from global into LDS at lds_addr.
// Descriptor per CDNA5 ISA ch.8: group0 = {count=1, lds_addr, global_addr,
// type=2}; group1 = {data_size=2B, tensor dims, tile dims, dim0 stride}.
// Tracked by TENSORcnt.  Issue from a single (uniform) wave.
DEV void tdm_load_2d(unsigned lds_addr, const void* gaddr,
                     unsigned tensor_d0, unsigned tensor_d1,
                     unsigned tile_d0, unsigned tile_d1, unsigned stride0) {
  unsigned long long ga = (unsigned long long)(uintptr_t)gaddr;
  U32x4 g0;
  g0.w[0] = 1u;                                   // count=1, user descriptor
  g0.w[1] = lds_addr;                             // LDS byte address
  g0.w[2] = (unsigned)ga;                         // global_addr[31:0]
  g0.w[3] = (unsigned)(ga >> 32) | (2u << 30);    // global_addr[56:32] | type=2
  U32x8 g1;
  g1.w[0] = 1u << 16;                             // data_size = 1 -> 2 bytes
  g1.w[1] = (tensor_d0 & 0xffffu) << 16;          // tensor_dim0[15:0]
  g1.w[2] = (tensor_d0 >> 16) | ((tensor_d1 & 0xffffu) << 16);
  g1.w[3] = (tensor_d1 >> 16) | (tile_d0 << 16);  // tensor_dim1 hi | tile_dim0
  g1.w[4] = tile_d1;                              // tile_dim1 (tile_dim2 = 0)
  g1.w[5] = stride0;                              // tensor_dim0_stride[31:0]
  g1.w[6] = 0u;
  g1.w[7] = 0u;
  U32x4 gz2; gz2.w[0] = 0; gz2.w[1] = 0; gz2.w[2] = 0; gz2.w[3] = 0;
  U32x4 gz3; gz3.w[0] = 0; gz3.w[1] = 0; gz3.w[2] = 0; gz3.w[3] = 0;
  asm volatile("tensor_load_to_lds %0, %1, %2, %3"
               :: "s"(g0.v), "s"(g1.v), "s"(gz2.v), "s"(gz3.v)
               : "memory");
}

// ---------------------------------------------------------------------------
// h = x  (fp32 copy into d_out)
// ---------------------------------------------------------------------------
__global__ __launch_bounds__(256) void k_copy4(const float4* __restrict__ in,
                                               float4* __restrict__ out, int n4) {
  int i = blockIdx.x * 256 + threadIdx.x;
  if (i < n4) out[i] = in[i];
}

// ---------------------------------------------------------------------------
// Weight convert+transpose: fp32 [D,K,N] -> bf16 [D,N,K]
// ---------------------------------------------------------------------------
__global__ __launch_bounds__(256) void k_wt(const float* __restrict__ w,
                                            unsigned short* __restrict__ wt,
                                            int K, int N, int total) {
  int i = blockIdx.x * 256 + threadIdx.x;
  if (i >= total) return;
  int kn = K * N;
  int d  = i / kn;
  int rem = i - d * kn;
  int k = rem / N;
  int n = rem - k * N;
  wt[(size_t)d * kn + (size_t)n * K + k] = f2bf(w[i]);
}

// ---------------------------------------------------------------------------
// LayerNorm: fp32 [TOK,768] -> bf16 [TOK,768].  One block per token.
// ---------------------------------------------------------------------------
__global__ __launch_bounds__(256) void k_layernorm(const float* __restrict__ x,
                                                   const float* __restrict__ g,
                                                   const float* __restrict__ b,
                                                   unsigned short* __restrict__ y) {
  __shared__ float red[256];
  int tok = blockIdx.x;
  int t   = threadIdx.x;
  const float* xr = x + (size_t)tok * TDIM;
  float v0 = xr[t], v1 = xr[t + 256], v2 = xr[t + 512];

  red[t] = v0 + v1 + v2;
  __syncthreads();
  for (int o = 128; o > 0; o >>= 1) {
    if (t < o) red[t] += red[t + o];
    __syncthreads();
  }
  float mu = red[0] * (1.0f / TDIM);
  __syncthreads();

  float d0 = v0 - mu, d1 = v1 - mu, d2 = v2 - mu;
  red[t] = d0 * d0 + d1 * d1 + d2 * d2;
  __syncthreads();
  for (int o = 128; o > 0; o >>= 1) {
    if (t < o) red[t] += red[t + o];
    __syncthreads();
  }
  float rstd = rsqrtf(red[0] * (1.0f / TDIM) + 1e-5f);

  unsigned short* yr = y + (size_t)tok * TDIM;
  yr[t]       = f2bf(d0 * rstd * g[t]       + b[t]);
  yr[t + 256] = f2bf(d1 * rstd * g[t + 256] + b[t + 256]);
  yr[t + 512] = f2bf(d2 * rstd * g[t + 512] + b[t + 512]);
}

// ---------------------------------------------------------------------------
// GEMM: C[M,N] = A[M,K](bf16) * Bt[N,K](bf16)^T  (+bias, +gelu, +residual)
// 256 threads = 8 waves; 128x256 block tile; each wave owns a 64x64 output
// tile (16 WMMA per 32-deep K step).  A/B tiles double-buffered in LDS and
// filled with global_load_async_to_lds_b128; 6 async instrs per thread/tile,
// so the pipelined wait is s_wait_asynccnt<=6.
// ---------------------------------------------------------------------------
template <int BIAS, int RESID, int GELU_, int OUTBF>
__global__ __launch_bounds__(256) void k_gemm(const unsigned short* __restrict__ A,
                                              const unsigned short* __restrict__ Bt,
                                              const float* __restrict__ bias,
                                              float* __restrict__ Cf,
                                              unsigned short* __restrict__ Obf,
                                              int M, int N, int K) {
  __shared__ __align__(16) unsigned short As[2][128 * 32];
  __shared__ __align__(16) unsigned short Bs[2][256 * 32];

  int tid  = threadIdx.x;
  int lane = tid & 31;
  int wave = tid >> 5;
  int wm   = wave & 1;   // 0..1 -> 64-row stripe
  int wn   = wave >> 1;  // 0..3 -> 64-col stripe
  int m0   = blockIdx.y * 128;
  int n0   = blockIdx.x * 256;

  v8f c[4][4];
#pragma unroll
  for (int mt = 0; mt < 4; ++mt)
#pragma unroll
    for (int nt = 0; nt < 4; ++nt) c[mt][nt] = (v8f){};

  // Per-thread staging slice: A: 16 elems of one row; B: one full 32-elem row.
  int arow = tid >> 1;
  int acol = (tid & 1) * 16;
  const unsigned short* gA = A  + (size_t)(m0 + arow) * K + acol;
  const unsigned short* gB = Bt + (size_t)(n0 + tid) * K;
  unsigned asb[2] = { (unsigned)(uintptr_t)&As[0][0], (unsigned)(uintptr_t)&As[1][0] };
  unsigned bsb[2] = { (unsigned)(uintptr_t)&Bs[0][0], (unsigned)(uintptr_t)&Bs[1][0] };
  unsigned aoff = (unsigned)(arow * 32 + acol) * 2;
  unsigned boff = (unsigned)(tid * 32) * 2;

  int nk = K / 32;

  // Prologue: stage tile 0 into buffer 0.
  async_lds_b128(asb[0] + aoff,      gA);
  async_lds_b128(asb[0] + aoff + 16, gA + 8);
#pragma unroll
  for (int j = 0; j < 4; ++j) async_lds_b128(bsb[0] + boff + j * 16, gB + j * 8);

  for (int kt = 0; kt < nk; ++kt) {
    int buf = kt & 1;
    __syncthreads();  // all waves done computing from buf^1 (prev iteration)
    if (kt + 1 < nk) {
      const unsigned short* pa = gA + (kt + 1) * 32;
      const unsigned short* pb = gB + (kt + 1) * 32;
      async_lds_b128(asb[buf ^ 1] + aoff,      pa);
      async_lds_b128(asb[buf ^ 1] + aoff + 16, pa + 8);
#pragma unroll
      for (int j = 0; j < 4; ++j)
        async_lds_b128(bsb[buf ^ 1] + boff + j * 16, pb + j * 8);
      asm volatile("s_wait_asynccnt 0x6" ::: "memory");  // tile kt landed
    } else {
      asm volatile("s_wait_asynccnt 0x0" ::: "memory");
    }
    __syncthreads();  // tile kt visible to all waves

    const unsigned short* as0 = &As[buf][0];
    const unsigned short* bs0 = &Bs[buf][0];
    v16bf af[4], bfr[4];
#pragma unroll
    for (int mt = 0; mt < 4; ++mt)
      af[mt] = load_frag16x32(as0 + (wm * 64 + mt * 16) * 32, 32, 0);
#pragma unroll
    for (int nt = 0; nt < 4; ++nt)
      bfr[nt] = load_frag16x32(bs0 + (wn * 64 + nt * 16) * 32, 32, 0);
#pragma unroll
    for (int mt = 0; mt < 4; ++mt)
#pragma unroll
      for (int nt = 0; nt < 4; ++nt)
        c[mt][nt] = wmma_bf16(af[mt], bfr[nt], c[mt][nt]);
  }

  // Epilogue.  C-frag layout: VGPR r -> row r (lanes 0-15) / row r+8 (16-31)
  int halfrow = (lane >> 4) << 3;
#pragma unroll
  for (int mt = 0; mt < 4; ++mt)
#pragma unroll
    for (int nt = 0; nt < 4; ++nt) {
      int col = n0 + wn * 64 + nt * 16 + (lane & 15);
#pragma unroll
      for (int r = 0; r < 8; ++r) {
        int row = m0 + wm * 64 + mt * 16 + r + halfrow;
        float v = c[mt][nt][r];
        if (BIAS)  v += bias[col];
        if (GELU_) v  = gelu_tanh(v);
        size_t idx = (size_t)row * N + col;
        if (RESID) v += Cf[idx];
        if (OUTBF) Obf[idx] = f2bf(v);
        else       Cf[idx]  = v;
      }
    }
}

// ---------------------------------------------------------------------------
// Flash-style attention.  Grid: (N/128, HEADS, B), 256 threads = 8 waves.
// Q (128x64) and K (64x64) tiles staged by the Tensor Data Mover; V staged
// transposed by VALU threads (TDM cannot transpose).  Online softmax.
// qkv: bf16 [TOK, 2304] (q|k|v);  out: bf16 [TOK, 768]
// ---------------------------------------------------------------------------
__global__ __launch_bounds__(256) void k_attn(const unsigned short* __restrict__ qkv,
                                              unsigned short* __restrict__ out) {
  __shared__ __align__(16) unsigned short Qs [128 * 64];   // [qrow][dh]
  __shared__ __align__(16) unsigned short Ks [ 64 * 64];   // [key][dh]
  __shared__ __align__(16) unsigned short Vts[ 64 * 64];   // [dh][key]
  __shared__ __align__(16) unsigned short Ps [8 * 16 * 64];// per-wave P tile

  int qb = blockIdx.x, h = blockIdx.y, b = blockIdx.z;
  int tid = threadIdx.x, lane = tid & 31, wave = tid >> 5;
  size_t base = (size_t)b * TN * QKVW;
  int qcol = h * TDH, kcol = TDIM + h * TDH, vcol = 2 * TDIM + h * TDH;

  // Stage Q block (128x64) via TDM: one descriptor, rows stride QKVW elems.
  if (wave == 0) {
    tdm_load_2d((unsigned)(uintptr_t)&Qs[0],
                qkv + base + (size_t)(qb * 128) * QKVW + qcol,
                QKVW, TOK, TDH, 128, QKVW);
    __builtin_amdgcn_s_wait_tensorcnt(0);
  }

  float mstate[8], lstate[8];
  v8f o[4];
#pragma unroll
  for (int r = 0; r < 8; ++r) { mstate[r] = -3.402823466e38f; lstate[r] = 0.0f; }
#pragma unroll
  for (int dt = 0; dt < 4; ++dt) o[dt] = (v8f){};

  unsigned short* Pw = Ps + wave * 16 * 64;
  const float scale = 0.125f;  // 1/sqrt(64)

  for (int kv0 = 0; kv0 < TN; kv0 += 64) {
    __syncthreads();  // protect Ks/Vts from previous iteration's consumers
    // Stage K (64x64) via TDM (wave 0), V transposed via all threads.
    if (wave == 0) {
      tdm_load_2d((unsigned)(uintptr_t)&Ks[0],
                  qkv + base + (size_t)kv0 * QKVW + kcol,
                  QKVW, TOK, TDH, 64, QKVW);
    }
#pragma unroll 4
    for (int i = 0; i < 16; ++i) {
      int e = tid * 16 + i;
      int key = e >> 6, dh = e & 63;
      Vts[dh * 64 + key] = qkv[base + (size_t)(kv0 + key) * QKVW + vcol + dh];
    }
    if (kv0 + 64 < TN)  // gfx1250 global_prefetch of next KV block (V rows)
      __builtin_prefetch(qkv + base + (size_t)(kv0 + 64 + (tid >> 2)) * QKVW + vcol, 0, 1);
    if (wave == 0) __builtin_amdgcn_s_wait_tensorcnt(0);
    __syncthreads();

    // S = Q * K^T  : 4 key tiles x (K=64 -> 2 WMMA steps)
    v8f sf[4];
#pragma unroll
    for (int nt = 0; nt < 4; ++nt) sf[nt] = (v8f){};
#pragma unroll
    for (int kk = 0; kk < 64; kk += 32) {
      v16bf aq = load_frag16x32(Qs + (wave * 16) * 64, 64, kk);
#pragma unroll
      for (int nt = 0; nt < 4; ++nt) {
        v16bf bk = load_frag16x32(Ks + nt * 16 * 64, 64, kk);
        sf[nt] = wmma_bf16(aq, bk, sf[nt]);
      }
    }

    // Online softmax (per C-frag row; half-wave shfl reductions)
    int halfrow = (lane >> 4) << 3;
#pragma unroll
    for (int r = 0; r < 8; ++r) {
      float mloc = -3.402823466e38f;
#pragma unroll
      for (int nt = 0; nt < 4; ++nt) {
        sf[nt][r] *= scale;
        mloc = fmaxf(mloc, sf[nt][r]);
      }
#pragma unroll
      for (int m = 1; m < 16; m <<= 1) mloc = fmaxf(mloc, __shfl_xor(mloc, m, 32));
      float mnew  = fmaxf(mstate[r], mloc);
      float alpha = __expf(mstate[r] - mnew);
      mstate[r]   = mnew;

      int prow = r + halfrow;
      float psum = 0.0f;
#pragma unroll
      for (int nt = 0; nt < 4; ++nt) {
        float p = __expf(sf[nt][r] - mnew);
        psum += p;
        Pw[prow * 64 + nt * 16 + (lane & 15)] = f2bf(p);
      }
#pragma unroll
      for (int m = 1; m < 16; m <<= 1) psum += __shfl_xor(psum, m, 32);
      lstate[r] = lstate[r] * alpha + psum;
#pragma unroll
      for (int dt = 0; dt < 4; ++dt) o[dt][r] *= alpha;
    }

    // O += P * V : 4 dh tiles x (K=64 keys -> 2 WMMA steps)
#pragma unroll
    for (int kk = 0; kk < 64; kk += 32) {
      v16bf ap = load_frag16x32(Pw, 64, kk);
#pragma unroll
      for (int dt = 0; dt < 4; ++dt) {
        v16bf bv = load_frag16x32(Vts + dt * 16 * 64, 64, kk);
        o[dt] = wmma_bf16(ap, bv, o[dt]);
      }
    }
  }

  // Write O / l  as bf16 [TOK, 768]
  int halfrow = (lane >> 4) << 3;
#pragma unroll
  for (int dt = 0; dt < 4; ++dt)
#pragma unroll
    for (int r = 0; r < 8; ++r) {
      int rowl  = wave * 16 + r + halfrow;
      int token = qb * 128 + rowl;
      float val = o[dt][r] / lstate[r];
      out[((size_t)b * TN + token) * TDIM + h * TDH + dt * 16 + (lane & 15)] = f2bf(val);
    }
}

// ---------------------------------------------------------------------------
// Host-side orchestration
// ---------------------------------------------------------------------------
extern "C" void kernel_launch(void* const* d_in, const int* in_sizes, int n_in,
                              void* d_out, int out_size, void* d_ws, size_t ws_size,
                              hipStream_t stream) {
  const float* x     = (const float*)d_in[0];
  const float* ln1_g = (const float*)d_in[1];
  const float* ln1_b = (const float*)d_in[2];
  const float* w_qkv = (const float*)d_in[3];
  const float* w_out = (const float*)d_in[4];
  const float* b_out = (const float*)d_in[5];
  const float* ln2_g = (const float*)d_in[6];
  const float* ln2_b = (const float*)d_in[7];
  const float* w1    = (const float*)d_in[8];
  const float* b1    = (const float*)d_in[9];
  const float* w2    = (const float*)d_in[10];
  const float* b2    = (const float*)d_in[11];

  float* h = (float*)d_out;  // fp32 hidden state [TOK, 768], residual accumulator

  // Workspace carve-up (bf16 buffers), 256B aligned
  char*  ws  = (char*)d_ws;
  size_t off = 0;
  auto carve = [&](size_t bytes) -> char* {
    char* p = ws + off;
    off += (bytes + 255) & ~(size_t)255;
    return p;
  };
  unsigned short* WQKVT = (unsigned short*)carve((size_t)TDEPTH * TDIM * QKVW * 2);
  unsigned short* WOUTT = (unsigned short*)carve((size_t)TDEPTH * TDIM * TDIM * 2);
  unsigned short* W1T   = (unsigned short*)carve((size_t)TDEPTH * TDIM * TMLP * 2);
  unsigned short* W2T   = (unsigned short*)carve((size_t)TDEPTH * TMLP * TDIM * 2);
  unsigned short* Y     = (unsigned short*)carve((size_t)TOK * TDIM * 2);
  unsigned short* QKV   = (unsigned short*)carve((size_t)TOK * QKVW * 2);
  unsigned short* AO    = (unsigned short*)carve((size_t)TOK * TDIM * 2);
  unsigned short* MID   = (unsigned short*)carve((size_t)TOK * TMLP * 2);
  (void)ws_size; (void)in_sizes; (void)n_in; (void)out_size;

  // --- Weight conversion (fp32 [D,K,N] -> bf16 [D,N,K]) ---
  {
    int t;
    t = TDEPTH * TDIM * QKVW;
    k_wt<<<(t + 255) / 256, 256, 0, stream>>>(w_qkv, WQKVT, TDIM, QKVW, t);
    t = TDEPTH * TDIM * TDIM;
    k_wt<<<(t + 255) / 256, 256, 0, stream>>>(w_out, WOUTT, TDIM, TDIM, t);
    t = TDEPTH * TDIM * TMLP;
    k_wt<<<(t + 255) / 256, 256, 0, stream>>>(w1, W1T, TDIM, TMLP, t);
    t = TDEPTH * TMLP * TDIM;
    k_wt<<<(t + 255) / 256, 256, 0, stream>>>(w2, W2T, TMLP, TDIM, t);
  }

  // --- h = x ---
  {
    int n4 = TOK * TDIM / 4;
    k_copy4<<<(n4 + 255) / 256, 256, 0, stream>>>((const float4*)x, (float4*)h, n4);
  }

  for (int l = 0; l < TDEPTH; ++l) {
    const unsigned short* wqkvt = WQKVT + (size_t)l * TDIM * QKVW;
    const unsigned short* woutt = WOUTT + (size_t)l * TDIM * TDIM;
    const unsigned short* w1t   = W1T   + (size_t)l * TDIM * TMLP;
    const unsigned short* w2t   = W2T   + (size_t)l * TMLP * TDIM;

    // LN1 -> Y(bf16)
    k_layernorm<<<TOK, 256, 0, stream>>>(h, ln1_g + l * TDIM, ln1_b + l * TDIM, Y);
    // QKV = Y @ Wqkv  (bf16 out, no bias)
    k_gemm<0, 0, 0, 1><<<dim3(QKVW / 256, TOK / 128), 256, 0, stream>>>(
        Y, wqkvt, nullptr, nullptr, QKV, TOK, QKVW, TDIM);
    // Attention -> AO(bf16)
    k_attn<<<dim3(TN / 128, THEADS, TB), 256, 0, stream>>>(QKV, AO);
    // h += AO @ Wout + b_out
    k_gemm<1, 1, 0, 0><<<dim3(TDIM / 256, TOK / 128), 256, 0, stream>>>(
        AO, woutt, b_out + l * TDIM, h, nullptr, TOK, TDIM, TDIM);
    // LN2 -> Y
    k_layernorm<<<TOK, 256, 0, stream>>>(h, ln2_g + l * TDIM, ln2_b + l * TDIM, Y);
    // MID = gelu(Y @ W1 + b1)  (bf16 out)
    k_gemm<1, 0, 1, 1><<<dim3(TMLP / 256, TOK / 128), 256, 0, stream>>>(
        Y, w1t, b1 + l * TMLP, nullptr, MID, TOK, TMLP, TDIM);
    // h += MID @ W2 + b2
    k_gemm<1, 1, 0, 0><<<dim3(TDIM / 256, TOK / 128), 256, 0, stream>>>(
        MID, w2t, b2 + l * TDIM, h, nullptr, TOK, TDIM, TMLP);
  }
}
